// MultiHeadAttention_8564164788888
// MI455X (gfx1250) — matrix operand order repaired and checked by reading, not compiled, not run on hardware
//
#include <hip/hip_runtime.h>
#include <hip/hip_bf16.h>

// Problem constants
constexpr int Bc  = 4;
constexpr int Sc  = 2048;
constexpr int Dc  = 1024;
constexpr int Hc  = 16;
constexpr int DKc = 64;

typedef __attribute__((ext_vector_type(16))) __bf16 bf16x16;
typedef __attribute__((ext_vector_type(8)))  float  f32x8;

union FragU { bf16x16 v; uint4 q[2]; };

// ---------------------------------------------------------------------------
// Fragment loaders (wave32 WMMA layouts, CDNA5 ISA §7.12.2)
// A-matrix 16x32 bf16: lane m=l&15; half h=l>>4; K run [h*16, h*16+16)
// contiguous (same k order as B) -> two 16B loads.
// B-matrix 32x16 bf16: lane n=l&15; half h=l>>4; K run [h*16, h*16+16)
// contiguous -> two 16B loads. Row-major source with row stride `ld`.
// ---------------------------------------------------------------------------
__device__ __forceinline__ bf16x16 load_fragA_g(const __bf16* p, size_t row_base,
                                                int ld, int kb) {
  int lane = threadIdx.x & 31;
  const __bf16* base =
      p + (row_base + (size_t)(lane & 15)) * (size_t)ld + kb + ((lane >> 4) << 4);
  FragU u;
  u.q[0] = *(const uint4*)(base);
  u.q[1] = *(const uint4*)(base + 8);
  return u.v;
}

__device__ __forceinline__ bf16x16 load_fragB_g(const __bf16* p, size_t row_base,
                                                int ld, int kb) {
  int lane = threadIdx.x & 31;
  const __bf16* base =
      p + (row_base + (size_t)(lane & 15)) * (size_t)ld + kb + ((lane >> 4) << 4);
  FragU u;
  u.q[0] = *(const uint4*)(base);
  u.q[1] = *(const uint4*)(base + 8);
  return u.v;
}

__device__ __forceinline__ f32x8 wmma_bf16(bf16x16 a, bf16x16 b, f32x8 c) {
  return __builtin_amdgcn_wmma_f32_16x16x32_bf16(
      /*neg_a=*/false, a, /*neg_b=*/false, b,
      /*c_mod=*/(short)0, c, /*reuse_a=*/false, /*reuse_b=*/false);
}

// Low 32 bits of a generic pointer into __shared__ == LDS byte offset
__device__ __forceinline__ unsigned lds_off(const void* p) {
  return (unsigned)(unsigned long long)p;
}

__device__ __forceinline__ void async_copy_b128(unsigned ldsaddr,
                                                const void* gaddr) {
  asm volatile("global_load_async_to_lds_b128 %0, %1, off"
               :: "v"(ldsaddr), "v"((unsigned long long)gaddr)
               : "memory");
}

__device__ __forceinline__ void wait_async0() {
  asm volatile("s_wait_asynccnt 0" ::: "memory");
}

// ---------------------------------------------------------------------------
// fp32 -> bf16 conversion
// ---------------------------------------------------------------------------
__global__ void cvt_f32_bf16(const float* __restrict__ in, __bf16* __restrict__ out,
                             int n) {
  int i = blockIdx.x * blockDim.x + threadIdx.x;
  if (i < n) out[i] = (__bf16)in[i];
}

// ---------------------------------------------------------------------------
// NT GEMM: Y[8192 x 1024] = A * W^T + bias.
// Block = 4 waves = 64x64 output tile. Per K=32 step, the 64x32 W tile is
// staged into LDS with async-to-LDS DMA and shared by all 4 waves (4x less
// global B traffic); A fragments are private per wave, loaded from global.
// mode 0: write bf16 as Q/K layout  [B, H, S, 64]
// mode 1: write bf16 as V^T layout  [B, H, 64, S]
// mode 2: write fp32 row-major      [M, 1024]
// ---------------------------------------------------------------------------
__global__ void gemm_nt_kernel(const __bf16* __restrict__ A,
                               const __bf16* __restrict__ W,
                               const float* __restrict__ bias,
                               __bf16* __restrict__ outb,
                               float* __restrict__ outf, int mode) {
  __shared__ __align__(16) __bf16 wtile[64 * 32];   // 4 KB W tile, row-major

  int tid  = threadIdx.x;
  int lane = tid & 31;
  int wave = tid >> 5;
  int nt64 = blockIdx.x & 15;            // 1024/64 = 16 column tiles
  int mt   = blockIdx.x >> 4;            // 8192/64 = 128 row tiles
  size_t row_base = (size_t)mt * 64 + wave * 16;
  int nbase = nt64 * 64;

  // staging: 256 16B chunks; thread t copies chunks 2t and 2t+1 (32B, one row
  // holds 4 chunks of its 64B). chunk c -> W row c>>2, k-offset (c&3)*8 elems.
  int c0 = tid * 2, c1 = c0 + 1;
  int sr0 = c0 >> 2, sk0 = (c0 & 3) * 8;
  int sr1 = c1 >> 2, sk1 = (c1 & 3) * 8;
  const __bf16* g0 = W + (size_t)(nbase + sr0) * Dc + sk0;
  const __bf16* g1 = W + (size_t)(nbase + sr1) * Dc + sk1;
  unsigned l0 = lds_off(&wtile[sr0 * 32 + sk0]);
  unsigned l1 = lds_off(&wtile[sr1 * 32 + sk1]);

  f32x8 acc[4];
#pragma unroll
  for (int j = 0; j < 4; ++j)
#pragma unroll
    for (int i = 0; i < 8; ++i) acc[j][i] = 0.0f;

  for (int kb = 0; kb < Dc; kb += 32) {
    // ---- async DMA the W tile into LDS
    async_copy_b128(l0, g0 + kb);
    async_copy_b128(l1, g1 + kb);
    wait_async0();
    __syncthreads();

    // ---- private A fragment from global
    bf16x16 a = load_fragA_g(A, row_base, Dc, kb);

    // ---- B fragments from LDS (lane n = l&15, K run (l>>4)*16, contiguous)
#pragma unroll
    for (int j = 0; j < 4; ++j) {
      const __bf16* bbase = &wtile[(j * 16 + (lane & 15)) * 32 + ((lane >> 4) << 4)];
      FragU u;
      u.q[0] = *(const uint4*)(bbase);
      u.q[1] = *(const uint4*)(bbase + 8);
      acc[j] = wmma_bf16(a, u.v, acc[j]);
    }
    __syncthreads();
  }

  int nloc = lane & 15;
  int mloc = (lane >> 4) << 3;           // C layout: row = mloc + r, col = nloc
#pragma unroll
  for (int j = 0; j < 4; ++j) {
    int col = nbase + j * 16 + nloc;
    float bv = bias[col];
#pragma unroll
    for (int r = 0; r < 8; ++r) {
      size_t m = row_base + mloc + r;    // global row = b*S + s
      float v = acc[j][r] + bv;
      if (mode == 2) {
        outf[m * (size_t)Dc + col] = v;
      } else {
        size_t b = m >> 11;              // / 2048
        size_t s = m & 2047;
        int hh = col >> 6;               // head
        int dd = col & 63;               // dim in head
        size_t idx = (mode == 0)
                         ? (((b * Hc + hh) * (size_t)Sc + s) * DKc + dd)
                         : (((b * Hc + hh) * (size_t)DKc + dd) * Sc + s);
        outb[idx] = (__bf16)v;
      }
    }
  }
}

// ---------------------------------------------------------------------------
// Flash attention: one wave (32 threads) per 16-row q tile per (b,h).
// Q,K: [B,H,S,64] bf16; Vt: [B,H,64,S] bf16; vals out: [B,S,1024] bf16.
// ---------------------------------------------------------------------------
__global__ void attn_kernel(const __bf16* __restrict__ Q,
                            const __bf16* __restrict__ K,
                            const __bf16* __restrict__ Vt,
                            __bf16* __restrict__ vals) {
  __shared__ __align__(16) __bf16 pb[16 * 32];   // P tile round-trip buffer

  int lane = threadIdx.x;
  int qt = blockIdx.x & 127;   // S/16 = 128
  int bh = blockIdx.x >> 7;    // 0..63
  int b = bh >> 4, h = bh & 15;
  size_t qbase = (size_t)qt * 16;

  const __bf16* Qp = Q + (size_t)bh * Sc * DKc;
  const __bf16* Kp = K + (size_t)bh * Sc * DKc;
  const __bf16* Vp = Vt + (size_t)bh * DKc * Sc;

  bf16x16 qa0 = load_fragA_g(Qp, qbase, DKc, 0);
  bf16x16 qa1 = load_fragA_g(Qp, qbase, DKc, 32);

  f32x8 O[4];
#pragma unroll
  for (int t = 0; t < 4; ++t)
#pragma unroll
    for (int i = 0; i < 8; ++i) O[t][i] = 0.0f;

  float rmax[8], rsum[8];
#pragma unroll
  for (int r = 0; r < 8; ++r) { rmax[r] = -1.0e30f; rsum[r] = 0.0f; }

  int nloc = lane & 15;
  int mloc = (lane >> 4) << 3;
  int hh8  = (lane >> 4) << 3;
  int mA   = lane & 15;

  int nkb = ((int)qbase + 16 + 31) >> 5;   // causal: only blocks up to the diag
  for (int ib = 0; ib < nkb; ++ib) {
    int kb = ib << 5;

    // ---- S = Q * K^T (two 16-key column tiles, K-dim 64 = 2 chained wmma)
    f32x8 sf[2];
#pragma unroll
    for (int ft = 0; ft < 2; ++ft) {
      bf16x16 b0 = load_fragB_g(Kp, (size_t)(kb + ft * 16), DKc, 0);
      bf16x16 b1 = load_fragB_g(Kp, (size_t)(kb + ft * 16), DKc, 32);
      f32x8 c;
#pragma unroll
      for (int i = 0; i < 8; ++i) c[i] = 0.0f;
      c = wmma_bf16(qa0, b0, c);
      c = wmma_bf16(qa1, b1, c);
      sf[ft] = c;
    }

    // ---- scale + causal mask + tile row max
    float tmax[8];
#pragma unroll
    for (int r = 0; r < 8; ++r) {
      int m = (int)qbase + mloc + r;
      float s0 = sf[0][r] * 0.125f;          // 1/sqrt(64)
      float s1 = sf[1][r] * 0.125f;
      if (kb + nloc > m)      s0 = -1.0e9f;
      if (kb + 16 + nloc > m) s1 = -1.0e9f;
      sf[0][r] = s0; sf[1][r] = s1;
      tmax[r] = fmaxf(s0, s1);
    }
#pragma unroll
    for (int mk = 1; mk <= 8; mk <<= 1)
#pragma unroll
      for (int r = 0; r < 8; ++r)
        tmax[r] = fmaxf(tmax[r], __shfl_xor(tmax[r], mk, 32));

    // ---- online softmax update
    float corr[8], psum[8];
#pragma unroll
    for (int r = 0; r < 8; ++r) {
      float nm = fmaxf(rmax[r], tmax[r]);
      corr[r] = __expf(rmax[r] - nm);
      rmax[r] = nm;
      float p0 = __expf(sf[0][r] - nm);
      float p1 = __expf(sf[1][r] - nm);
      sf[0][r] = p0; sf[1][r] = p1;
      psum[r] = p0 + p1;
    }
#pragma unroll
    for (int mk = 1; mk <= 8; mk <<= 1)
#pragma unroll
      for (int r = 0; r < 8; ++r)
        psum[r] += __shfl_xor(psum[r], mk, 32);
#pragma unroll
    for (int r = 0; r < 8; ++r) rsum[r] = rsum[r] * corr[r] + psum[r];
#pragma unroll
    for (int t = 0; t < 4; ++t)
#pragma unroll
      for (int r = 0; r < 8; ++r) O[t][r] *= corr[r];

    // ---- P: C-layout f32 -> LDS row-major bf16 -> A-layout fragment
#pragma unroll
    for (int r = 0; r < 8; ++r) {
      pb[(mloc + r) * 32 + nloc]      = (__bf16)sf[0][r];
      pb[(mloc + r) * 32 + 16 + nloc] = (__bf16)sf[1][r];
    }
    asm volatile("s_wait_dscnt 0" ::: "memory");
    __builtin_amdgcn_wave_barrier();
    FragU up;
    up.q[0] = *(const uint4*)(&pb[mA * 32 + 2 * hh8]);
    up.q[1] = *(const uint4*)(&pb[mA * 32 + 2 * hh8 + 8]);
    bf16x16 pA = up.v;
    __builtin_amdgcn_wave_barrier();
    asm volatile("" ::: "memory");

    // ---- O += P * V  (B operand contiguous thanks to V^T layout)
#pragma unroll
    for (int nt = 0; nt < 4; ++nt) {
      bf16x16 vb = load_fragB_g(Vp, (size_t)(nt * 16), Sc, kb);
      O[nt] = wmma_bf16(pA, vb, O[nt]);
    }
  }

  // ---- normalize and write vals[b, s, h*64 + d] as bf16
  float rinv[8];
#pragma unroll
  for (int r = 0; r < 8; ++r) rinv[r] = 1.0f / rsum[r];
#pragma unroll
  for (int nt = 0; nt < 4; ++nt)
#pragma unroll
    for (int r = 0; r < 8; ++r) {
      size_t srow = (size_t)b * Sc + qbase + mloc + r;
      size_t col  = (size_t)h * DKc + nt * 16 + nloc;
      vals[srow * (size_t)Dc + col] = (__bf16)(O[nt][r] * rinv[r]);
    }
}

// ---------------------------------------------------------------------------
// Host launcher
// ---------------------------------------------------------------------------
extern "C" void kernel_launch(void* const* d_in, const int* in_sizes, int n_in,
                              void* d_out, int out_size, void* d_ws, size_t ws_size,
                              hipStream_t stream) {
  const float* x  = (const float*)d_in[0];
  // d_in[1] = mask (causal, computed analytically -> unused)
  const float* Wq = (const float*)d_in[2];
  const float* bq = (const float*)d_in[3];
  const float* Wk = (const float*)d_in[4];
  const float* bk = (const float*)d_in[5];
  const float* Wv = (const float*)d_in[6];
  const float* bv = (const float*)d_in[7];
  const float* Wo = (const float*)d_in[8];
  const float* bo = (const float*)d_in[9];

  const size_t NX = (size_t)Bc * Sc * Dc;   // 8,388,608
  const size_t NW = (size_t)Dc * Dc;        // 1,048,576

  char* ws = (char*)d_ws;
  __bf16* xb   = (__bf16*)ws;            ws += NX * 2;
  __bf16* wqb  = (__bf16*)ws;            ws += NW * 2;
  __bf16* wkb  = (__bf16*)ws;            ws += NW * 2;
  __bf16* wvb  = (__bf16*)ws;            ws += NW * 2;
  __bf16* wob  = (__bf16*)ws;            ws += NW * 2;
  __bf16* Qb   = (__bf16*)ws;            ws += NX * 2;
  __bf16* Kb   = (__bf16*)ws;            ws += NX * 2;
  __bf16* Vtb  = (__bf16*)ws;            ws += NX * 2;
  __bf16* valb = (__bf16*)ws;            ws += NX * 2;   // ~92 MB total

  // 1) convert inputs to bf16
  cvt_f32_bf16<<<(int)((NX + 255) / 256), 256, 0, stream>>>(x, xb, (int)NX);
  cvt_f32_bf16<<<(int)((NW + 255) / 256), 256, 0, stream>>>(Wq, wqb, (int)NW);
  cvt_f32_bf16<<<(int)((NW + 255) / 256), 256, 0, stream>>>(Wk, wkb, (int)NW);
  cvt_f32_bf16<<<(int)((NW + 255) / 256), 256, 0, stream>>>(Wv, wvb, (int)NW);
  cvt_f32_bf16<<<(int)((NW + 255) / 256), 256, 0, stream>>>(Wo, wob, (int)NW);

  // 2) QKV projections (WMMA NT GEMM, LDS-staged W).  M=8192 rows, N=1024.
  dim3 ggrid(((Bc * Sc) / 64) * (Dc / 64));    // 128 * 16 = 2048 blocks
  dim3 gblk(128);                              // 4 waves
  gemm_nt_kernel<<<ggrid, gblk, 0, stream>>>(xb, wqb, bq, Qb, nullptr, 0);
  gemm_nt_kernel<<<ggrid, gblk, 0, stream>>>(xb, wkb, bk, Kb, nullptr, 0);
  gemm_nt_kernel<<<ggrid, gblk, 0, stream>>>(xb, wvb, bv, Vtb, nullptr, 1);

  // 3) causal flash attention, one wave per 16 q-rows per (b,h)
  dim3 agrid(Bc * Hc * (Sc / 16));             // 8192
  attn_kernel<<<agrid, 32, 0, stream>>>(Qb, Kb, Vtb, valb);

  // 4) output projection -> fp32 d_out
  gemm_nt_kernel<<<ggrid, gblk, 0, stream>>>(valb, wob, bo, nullptr, (float*)d_out, 2);
}
